// RNN_V3_22763326669286
// MI455X (gfx1250) — matrix operand order repaired and checked by reading, not compiled, not run on hardware
//
#include <hip/hip_runtime.h>

typedef _Float16 h16;
typedef __attribute__((ext_vector_type(16))) _Float16 v16h;
typedef __attribute__((ext_vector_type(8)))  _Float16 v8h;
typedef __attribute__((ext_vector_type(8)))  float    v8f;
typedef __attribute__((ext_vector_type(4)))  float    f32x4;

#define LN_EPS 1e-5f

__device__ __forceinline__ v16h cat8(v8h lo, v8h hi) {
  return __builtin_shufflevector(lo, hi, 0,1,2,3,4,5,6,7,8,9,10,11,12,13,14,15);
}

// ---------------------------------------------------------------------------
// WMMA GEMM: C[M,N] = A[M,K] @ W[N,K]^T (+bias[n]) (+addB[(m%256),n]) (relu?)
// A, W f16 row-major. Compile-time epilogue (branch-free). Double-buffered LDS,
// K staged in 64-wide chunks (one barrier per stage). Block=128 (4 waves),
// WG tile 64x64, wave tile 32x32 = 2x2 v_wmma_f32_16x16x32_f16 per k-step.
// M,N multiples of 64; K multiple of 64.
// ---------------------------------------------------------------------------
template<bool BIAS, bool ADD, bool RELU, bool OUTF, bool OUTH>
__global__ __launch_bounds__(128)
void gemm16(const h16* __restrict__ A, int lda,
            const h16* __restrict__ W, int ldw,
            const float* __restrict__ bias,
            const float* __restrict__ addB, int addLd,
            float* __restrict__ Cf, h16* __restrict__ Ch,
            int N, int K)
{
  __shared__ __align__(16) h16 As[2][64 * 72];
  __shared__ __align__(16) h16 Bs[2][64 * 72];

  const int tid  = threadIdx.x;
  const int m0   = blockIdx.y * 64;
  const int n0   = blockIdx.x * 64;
  const int lane = tid & 31;
  const int wv   = tid >> 5;
  const int wm   = (wv >> 1) * 32;
  const int wn   = (wv & 1) * 32;
  const int fr   = lane & 15;
  const int hiH  = (lane & 16) ? 1 : 0;
  const int koffA = hiH ? 8  : 0;   // A frag: lanes 16-31 hold K=8..15 / 24..31
  const int koffB = hiH ? 8  : 0;   // B frag: lanes 16-31 hold K=8..15 / 24..31

  // staging: 512 chunks of 16B per 64x64-half tile, 4 per thread, coalesced
  int rowS[4], koS[4];
#pragma unroll
  for (int i = 0; i < 4; i++) { int c = tid + 128 * i; rowS[i] = c >> 3; koS[i] = (c & 7) * 8; }

  f32x4 ra[4], rb[4];
  auto loadStage = [&](int kc) {
#pragma unroll
    for (int i = 0; i < 4; i++) {
      ra[i] = *(const f32x4*)(A + (size_t)(m0 + rowS[i]) * lda + kc * 64 + koS[i]);
      rb[i] = *(const f32x4*)(W + (size_t)(n0 + rowS[i]) * ldw + kc * 64 + koS[i]);
    }
  };
  auto storeStage = [&](int b) {
#pragma unroll
    for (int i = 0; i < 4; i++) {
      *(f32x4*)(&As[b][rowS[i] * 72 + koS[i]]) = ra[i];
      *(f32x4*)(&Bs[b][rowS[i] * 72 + koS[i]]) = rb[i];
    }
  };

  v8f acc[2][2];
#pragma unroll
  for (int i = 0; i < 2; i++)
#pragma unroll
    for (int j = 0; j < 2; j++)
#pragma unroll
      for (int e = 0; e < 8; e++) acc[i][j][e] = 0.0f;

  loadStage(0);
  storeStage(0);
  const int nk = K >> 6;
  for (int kc = 0; kc < nk; kc++) {
    __syncthreads();                       // stores to buf (kc&1) visible
    if (kc + 1 < nk) loadStage(kc + 1);    // prefetch next stage into regs
    const h16* as = As[kc & 1];
    const h16* bs = Bs[kc & 1];
#pragma unroll
    for (int ks = 0; ks < 2; ks++) {       // two 32-wide k-steps per stage
      v16h af[2], bf[2];
#pragma unroll
      for (int i = 0; i < 2; i++) {
        int r = wm + 16 * i + fr;
        v8h lo = *(const v8h*)(&as[r * 72 + ks * 32 + koffA]);
        v8h hi = *(const v8h*)(&as[r * 72 + ks * 32 + koffA + 16]);
        af[i] = cat8(lo, hi);
      }
#pragma unroll
      for (int j = 0; j < 2; j++) {
        int r = wn + 16 * j + fr;
        v8h lo = *(const v8h*)(&bs[r * 72 + ks * 32 + koffB]);
        v8h hi = *(const v8h*)(&bs[r * 72 + ks * 32 + koffB + 16]);
        bf[j] = cat8(lo, hi);
      }
#pragma unroll
      for (int i = 0; i < 2; i++)
#pragma unroll
        for (int j = 0; j < 2; j++)
          acc[i][j] = __builtin_amdgcn_wmma_f32_16x16x32_f16(
              false, af[i], false, bf[j], (short)0, acc[i][j], false, false);
    }
    if (kc + 1 < nk) storeStage((kc + 1) & 1);  // opposite buffer: no race
  }

#pragma unroll
  for (int i = 0; i < 2; i++)
#pragma unroll
    for (int j = 0; j < 2; j++) {
      int col = n0 + wn + 16 * j + fr;
      float bv = 0.0f;
      if constexpr (BIAS) bv = bias[col];
#pragma unroll
      for (int vi = 0; vi < 8; vi++) {
        int rowm = m0 + wm + 16 * i + vi + (hiH ? 8 : 0);
        float v  = acc[i][j][vi] + bv;
        if constexpr (ADD)  v += addB[(size_t)(rowm & 255) * addLd + col];
        if constexpr (RELU) v = fmaxf(v, 0.0f);
        size_t o = (size_t)rowm * N + col;
        if constexpr (OUTF) Cf[o] = v;
        if constexpr (OUTH) Ch[o] = (h16)v;
      }
    }
}

// ---------------------------------------------------------------------------
// Elementwise / reduction kernels
// ---------------------------------------------------------------------------
__global__ void cvt_f32_f16(const float* __restrict__ s, h16* __restrict__ d, int n) {
  int i = blockIdx.x * 256 + threadIdx.x;
  if (i < n) d[i] = (h16)s[i];
}

__global__ void fill0f(float* p, int n) {
  int i = blockIdx.x * 256 + threadIdx.x;
  if (i < n) p[i] = 0.0f;
}
__global__ void fill0h(h16* p, int n) {
  int i = blockIdx.x * 256 + threadIdx.x;
  if (i < n) p[i] = (h16)0.0f;
}

// inference BatchNorm (feature on last axis, fmask = F-1, F power of two) -> f16
__global__ void bn_to_f16(const float* __restrict__ x, const float* __restrict__ g,
                          const float* __restrict__ b, const float* __restrict__ rm,
                          const float* __restrict__ rv, h16* __restrict__ y,
                          int n, int fmask) {
  int i = blockIdx.x * 256 + threadIdx.x;
  if (i < n) {
    int f = i & fmask;
    y[i] = (h16)((x[i] - rm[f]) * rsqrtf(rv[f] + LN_EPS) * g[f] + b[f]);
  }
}

// PyTorch GRUCell gate fusion. gi/gh: [B, 3H] f32, H=512.
__global__ void gru_update(const float* __restrict__ gi, const float* __restrict__ gh,
                           const float* __restrict__ bih, const float* __restrict__ bhh,
                           float* __restrict__ h, h16* __restrict__ h16o,
                           h16* __restrict__ hist) {
  int idx = blockIdx.x * 256 + threadIdx.x;   // B*H = 131072
  int b = idx >> 9, j = idx & 511;
  size_t base = (size_t)b * 1536;
  float ir  = gi[base + j]        + bih[j];
  float iz  = gi[base + 512 + j]  + bih[512 + j];
  float inn = gi[base + 1024 + j] + bih[1024 + j];
  float hr  = gh[base + j]        + bhh[j];
  float hz  = gh[base + 512 + j]  + bhh[512 + j];
  float hn  = gh[base + 1024 + j] + bhh[1024 + j];
  float r = 1.0f / (1.0f + expf(-(ir + hr)));
  float z = 1.0f / (1.0f + expf(-(iz + hz)));
  float n = tanhf(inn + r * hn);
  float out = (1.0f - z) * n + z * h[idx];
  h[idx]    = out;
  h16o[idx] = (h16)out;
  if (hist) hist[idx] = (h16)out;
}

// s = relu(LN_256(X)) + x_t   (one 256-thread block per row)
__global__ void ln_hp256(const float* __restrict__ X, const float* __restrict__ g,
                         const float* __restrict__ b, const h16* __restrict__ xt,
                         h16* __restrict__ sOut) {
  int row = blockIdx.x, tid = threadIdx.x;
  float v = X[(size_t)row * 256 + tid];
  __shared__ float red[256];
  red[tid] = v; __syncthreads();
  for (int s = 128; s > 0; s >>= 1) { if (tid < s) red[tid] += red[tid + s]; __syncthreads(); }
  float mu = red[0] * (1.0f / 256.0f);
  __syncthreads();
  float d = v - mu;
  red[tid] = d * d; __syncthreads();
  for (int s = 128; s > 0; s >>= 1) { if (tid < s) red[tid] += red[tid + s]; __syncthreads(); }
  float var = red[0] * (1.0f / 256.0f);
  float ln  = d * rsqrtf(var + LN_EPS) * g[tid] + b[tid];
  float hp  = fmaxf(ln, 0.0f);
  sOut[(size_t)row * 256 + tid] = (h16)(hp + (float)xt[(size_t)row * 256 + tid]);
}

// relu(LN_512(X)) over f16 rows, 256 threads / row, 2 elems per thread
__global__ void ln512_relu(const h16* __restrict__ X, const float* __restrict__ g,
                           const float* __restrict__ b, h16* __restrict__ Y) {
  int row = blockIdx.x, tid = threadIdx.x;
  const h16* x = X + (size_t)row * 512;
  float v0 = (float)x[tid], v1 = (float)x[tid + 256];
  __shared__ float red[256];
  red[tid] = v0 + v1; __syncthreads();
  for (int s = 128; s > 0; s >>= 1) { if (tid < s) red[tid] += red[tid + s]; __syncthreads(); }
  float mu = red[0] * (1.0f / 512.0f);
  __syncthreads();
  float d0 = v0 - mu, d1 = v1 - mu;
  red[tid] = d0 * d0 + d1 * d1; __syncthreads();
  for (int s = 128; s > 0; s >>= 1) { if (tid < s) red[tid] += red[tid + s]; __syncthreads(); }
  float rs = rsqrtf(red[0] * (1.0f / 512.0f) + LN_EPS);
  Y[(size_t)row * 512 + tid]       = (h16)fmaxf(d0 * rs * g[tid] + b[tid], 0.0f);
  Y[(size_t)row * 512 + tid + 256] = (h16)fmaxf(d1 * rs * g[tid + 256] + b[tid + 256], 0.0f);
}

// head l3: out[row*5 + head] = dot(Y[row, 0:512], w3) + b3   (one wave32 per row)
__global__ void head_l3(const h16* __restrict__ Y, const h16* __restrict__ w3,
                        const float* __restrict__ b3, float* __restrict__ out, int head) {
  int row  = blockIdx.x * 8 + (threadIdx.x >> 5);
  int lane = threadIdx.x & 31;
  const h16* y = Y + (size_t)row * 512;
  float s = 0.0f;
#pragma unroll
  for (int k = 0; k < 16; k++) s += (float)y[lane * 16 + k] * (float)w3[lane * 16 + k];
  for (int off = 16; off > 0; off >>= 1) s += __shfl_xor(s, off, 32);
  if (lane == 0) out[(size_t)row * 5 + head] = s + b3[0];
}

// ---------------------------------------------------------------------------
extern "C" void kernel_launch(void* const* d_in, const int* in_sizes, int n_in,
                              void* d_out, int out_size, void* d_ws, size_t ws_size,
                              hipStream_t stream) {
  (void)in_sizes; (void)n_in; (void)out_size; (void)ws_size;
  const int T = 256, B = 256, FFIX = 128, FSEQ = 256, H = 512;

  const float* Xfix = (const float*)d_in[0];
  const float* Xseq = (const float*)d_in[1];
  auto F = [&](int i) { return (const float*)d_in[i]; };

  // --- bump allocator over workspace ---
  char* cur = (char*)d_ws;
  auto aH = [&](size_t n) { h16* p = (h16*)cur;  cur += ((n * 2 + 255) & ~(size_t)255); return p; };
  auto aF = [&](size_t n) { float* p = (float*)cur; cur += ((n * 4 + 255) & ~(size_t)255); return p; };

  h16 *fcw1 = aH(512 * 128), *fcw2 = aH(512 * 512), *fcw3 = aH(512 * 512);
  h16 *g1Wih = aH(1536 * 256), *g1Whh = aH(1536 * 512);
  h16 *g4Wih = aH(1536 * 256), *g4Whh = aH(1536 * 512);
  h16 *iW1 = aH(512 * 512), *iW2 = aH(256 * 512);
  h16 *mW1 = aH(512 * 1024), *mW2 = aH(512 * 512), *mW3 = aH(512 * 512);
  h16 *hW1[5], *hW2[5], *hW3[5];
  for (int h = 0; h < 5; h++) { hW1[h] = aH(512 * 512); hW2[h] = aH(512 * 512); hW3[h] = aH(512); }

  h16* Xf     = aH((size_t)T * B * FSEQ);
  h16* Xfx    = aH((size_t)B * FFIX);
  h16* t1     = aH((size_t)B * H);
  h16* t2     = aH((size_t)B * H);
  h16* hfix16 = aH((size_t)B * H);
  h16* h1f16  = aH((size_t)B * H);
  h16* h4f16  = aH((size_t)B * H);
  h16* h4all  = aH((size_t)T * B * H);
  h16* y1     = aH((size_t)T * B * H);
  h16* y2     = aH((size_t)T * B * H);
  h16* outsB  = aH((size_t)T * B * H);
  float* mfix = aF((size_t)B * H);
  float* h1f  = aF((size_t)B * H);
  float* h4f  = aF((size_t)B * H);
  float* gb1  = aF((size_t)B * 1536);
  float* gb2  = aF((size_t)B * 1536);

  // --- weights -> f16 ---
  auto cvt = [&](int idx, h16* dst, int n) {
    cvt_f32_f16<<<(n + 255) / 256, 256, 0, stream>>>(F(idx), dst, n);
  };
  cvt(10, fcw1, 512 * 128); cvt(12, fcw2, 512 * 512); cvt(14, fcw3, 512 * 512);
  cvt(16, g1Wih, 1536 * 256); cvt(17, g1Whh, 1536 * 512);
  cvt(20, g4Wih, 1536 * 256); cvt(21, g4Whh, 1536 * 512);
  cvt(24, iW1, 512 * 512); cvt(26, iW2, 256 * 512);
  cvt(30, mW1, 512 * 1024); cvt(32, mW2, 512 * 512); cvt(34, mW3, 512 * 512);
  for (int h = 0; h < 5; h++) {
    int b0 = 36 + 8 * h;
    cvt(b0, hW1[h], 512 * 512); cvt(b0 + 4, hW2[h], 512 * 512); cvt(b0 + 6, hW3[h], 512);
  }

  // --- BatchNorm inputs -> f16 ---
  { int n = T * B * FSEQ;
    bn_to_f16<<<(n + 255) / 256, 256, 0, stream>>>(Xseq, F(6), F(7), F(8), F(9), Xf, n, FSEQ - 1); }
  { int n = B * FFIX;
    bn_to_f16<<<(n + 255) / 256, 256, 0, stream>>>(Xfix, F(2), F(3), F(4), F(5), Xfx, n, FFIX - 1); }

  // --- zero recurrent state (deterministic per call) ---
  { int n = B * H;
    fill0f<<<n / 256, 256, 0, stream>>>(h1f, n);
    fill0f<<<n / 256, 256, 0, stream>>>(h4f, n);
    fill0h<<<n / 256, 256, 0, stream>>>(h1f16, n);
    fill0h<<<n / 256, 256, 0, stream>>>(h4f16, n); }

  // dispatch to branch-free template instantiations
  auto gemm = [&](const h16* Ap, int lda, const h16* Wp, int ldw, const float* bias,
                  const float* addB, int addLd, float* Cf, h16* Ch,
                  int M, int N, int K, int relu) {
    dim3 g(N / 64, M / 64);
    if (addB) {                       // merge-l1: broadcast add + relu -> f16
      gemm16<false, true, true, false, true><<<g, 128, 0, stream>>>(
          Ap, lda, Wp, ldw, bias, addB, addLd, Cf, Ch, N, K);
    } else if (bias && relu) {        // bias + relu -> f16
      gemm16<true, false, true, false, true><<<g, 128, 0, stream>>>(
          Ap, lda, Wp, ldw, bias, addB, addLd, Cf, Ch, N, K);
    } else if (bias && Ch) {          // bias -> f16
      gemm16<true, false, false, false, true><<<g, 128, 0, stream>>>(
          Ap, lda, Wp, ldw, bias, addB, addLd, Cf, Ch, N, K);
    } else if (bias) {                // bias -> f32
      gemm16<true, false, false, true, false><<<g, 128, 0, stream>>>(
          Ap, lda, Wp, ldw, bias, addB, addLd, Cf, Ch, N, K);
    } else {                          // plain -> f32 (GRU gate GEMMs)
      gemm16<false, false, false, true, false><<<g, 128, 0, stream>>>(
          Ap, lda, Wp, ldw, bias, addB, addLd, Cf, Ch, N, K);
    }
  };

  // --- fc_fix MLP: h_fix [B,H] ---
  gemm(Xfx, FFIX, fcw1, FFIX, F(11), nullptr, 0, nullptr, t1, B, H, FFIX, 1);
  gemm(t1,  H,    fcw2, H,    F(13), nullptr, 0, nullptr, t2, B, H, H,    1);
  gemm(t2,  H,    fcw3, H,    F(15), nullptr, 0, nullptr, hfix16, B, H, H, 0);
  // mfix = h_fix @ merge.l1.W[:, :512]^T + b  (h_fix half of merge-l1, time-invariant)
  gemm(hfix16, H, mW1, 1024, F(31), nullptr, 0, mfix, nullptr, B, H, H, 0);

  // --- sequential scan (recurrent core only; merge/heads deferred) ---
  for (int t = 0; t < T; t++) {
    const h16* xt = Xf + (size_t)t * B * FSEQ;
    gemm(xt,    FSEQ, g1Wih, FSEQ, nullptr, nullptr, 0, gb1, nullptr, B, 3 * H, FSEQ, 0);
    gemm(h1f16, H,    g1Whh, H,    nullptr, nullptr, 0, gb2, nullptr, B, 3 * H, H,    0);
    gru_update<<<(B * H) / 256, 256, 0, stream>>>(gb1, gb2, F(18), F(19), h1f, h1f16, nullptr);
    gemm(h1f16, H, iW1, H, F(25), nullptr, 0, nullptr, t1, B, H, H, 1);
    gemm(t1,    H, iW2, H, F(27), nullptr, 0, gb1, nullptr, B, FSEQ, H, 0);
    ln_hp256<<<B, FSEQ, 0, stream>>>(gb1, F(28), F(29), xt, t2);
    gemm(t2,    FSEQ, g4Wih, FSEQ, nullptr, nullptr, 0, gb1, nullptr, B, 3 * H, FSEQ, 0);
    gemm(h4f16, H,    g4Whh, H,    nullptr, nullptr, 0, gb2, nullptr, B, 3 * H, H,    0);
    gru_update<<<(B * H) / 256, 256, 0, stream>>>(gb1, gb2, F(22), F(23), h4f, h4f16,
                                                  h4all + (size_t)t * B * H);
  }

  const int TB = T * B;  // 65536 rows, batched over all timesteps
  // --- merge MLP over [T*B, H] ---
  gemm(h4all, H, mW1 + 512, 1024, nullptr, mfix, H, nullptr, y1, TB, H, H, 1);
  gemm(y1,    H, mW2, H, F(33), nullptr, 0, nullptr, y2, TB, H, H, 1);
  gemm(y2,    H, mW3, H, F(35), nullptr, 0, nullptr, outsB, TB, H, H, 0);

  // --- 5 heads over [T*B, H] ---
  for (int h = 0; h < 5; h++) {
    int b0 = 36 + 8 * h;
    gemm(outsB, H, hW1[h], H, F(b0 + 1), nullptr, 0, nullptr, y1, TB, H, H, 0);
    ln512_relu<<<TB, 256, 0, stream>>>(y1, F(b0 + 2), F(b0 + 3), y2);
    gemm(y2, H, hW2[h], H, F(b0 + 5), nullptr, 0, nullptr, y1, TB, H, H, 1);
    head_l3<<<TB / 8, 256, 0, stream>>>(y1, hW3[h], F(b0 + 7), (float*)d_out, h);
  }
}
